// A2A_48515950576209
// MI455X (gfx1250) — matrix operand order repaired and checked by reading, not compiled, not run on hardware
//
#include <hip/hip_runtime.h>
#include <hip/hip_bf16.h>

#define DD 128
#define EPSV 1e-5f
#define NEGS 0.01f

typedef __attribute__((ext_vector_type(16))) __bf16 v16bf;
typedef __attribute__((ext_vector_type(8)))  __bf16 v8bf;
typedef __attribute__((ext_vector_type(8)))  float  v8f;

__device__ __forceinline__ __bf16 f2bf(float f) { return (__bf16)f; }

__device__ __forceinline__ v8f wmma_bf16(v16bf a, v16bf b, v8f c) {
    return __builtin_amdgcn_wmma_f32_16x16x32_bf16(
        false, a, false, b, (short)0, c, false, false);
}

__device__ __forceinline__ void zero8(v8f c[8]) {
#pragma unroll
    for (int ct = 0; ct < 8; ++ct)
#pragma unroll
        for (int v = 0; v < 8; ++v) c[ct][v] = 0.f;
}

// ---- bf16 A staging in LDS (row-major 16 x 128) ----------------------------
__device__ __forceinline__ void stage_rows(__bf16* dst, const float* src, int ld, int r0) {
    int lane = threadIdx.x & 31;
    int r = lane & 15, h = lane >> 4;
    const float* p = src + (size_t)(r0 + r) * ld + h * 64;
    __bf16* q = dst + r * DD + h * 64;
#pragma unroll
    for (int i = 0; i < 64; ++i) q[i] = f2bf(p[i]);
}

// A-matrix 16x32 bf16 operand from row-major bf16 LDS tile (ISA 7.12.2)
__device__ __forceinline__ v16bf load_a_lds(const __bf16* X, int k0) {
    int lane = threadIdx.x & 31;
    int m = lane & 15, h = lane >> 4;
    const __bf16* p = X + m * DD + k0 + h * 8;
    v8bf lo = *(const v8bf*)(p);
    v8bf hi = *(const v8bf*)(p + 16);
    return __builtin_shufflevector(lo, hi, 0, 1, 2, 3, 4, 5, 6, 7,
                                   8, 9, 10, 11, 12, 13, 14, 15);
}

// out(16 x 128) += Xlds(16 x 128 bf16) @ W^T  (W pre-packed bf16, B-tile order)
// Software-pipelined: prefetch kc+1's 8 B tiles while kc's WMMAs execute.
__device__ __forceinline__ void gemm16p(const __bf16* Xlds, const __bf16* P, v8f acc[8]) {
    int lane = threadIdx.x & 31;
    const v16bf* BP = (const v16bf*)P + lane;

    v16bf a0 = load_a_lds(Xlds, 0);
    v16bf a1 = load_a_lds(Xlds, 32);
    v16bf a2 = load_a_lds(Xlds, 64);
    v16bf a3 = load_a_lds(Xlds, 96);

    v16bf b[8], bn[8];
#pragma unroll
    for (int ct = 0; ct < 8; ++ct) b[ct] = BP[ct * 32];

#pragma unroll
    for (int kc = 0; kc < 4; ++kc) {
        v16bf a = (kc == 0) ? a0 : (kc == 1) ? a1 : (kc == 2) ? a2 : a3;
        if (kc < 3) {
#pragma unroll
            for (int ct = 0; ct < 8; ++ct) bn[ct] = BP[((kc + 1) * 8 + ct) * 32];
        }
#pragma unroll
        for (int ct = 0; ct < 8; ++ct) acc[ct] = wmma_bf16(a, b[ct], acc[ct]);
        if (kc < 3) {
#pragma unroll
            for (int ct = 0; ct < 8; ++ct) b[ct] = bn[ct];
        }
    }
}

// C/D layout: element (row m = v + 8*(lane/16), col j = ct*16 + lane%16)
__device__ __forceinline__ void store_cd(float* Y, int ld, int r0, const v8f c[8]) {
    int lane = threadIdx.x & 31;
    int n = lane & 15, h = lane >> 4;
#pragma unroll
    for (int ct = 0; ct < 8; ++ct)
#pragma unroll
        for (int v = 0; v < 8; ++v)
            Y[(size_t)(r0 + v + 8 * h) * ld + ct * 16 + n] = c[ct][v];
}

__device__ __forceinline__ void load_cd(const float* Y, int ld, int r0, v8f c[8]) {
    int lane = threadIdx.x & 31;
    int n = lane & 15, h = lane >> 4;
#pragma unroll
    for (int ct = 0; ct < 8; ++ct)
#pragma unroll
        for (int v = 0; v < 8; ++v)
            c[ct][v] = Y[(size_t)(r0 + v + 8 * h) * ld + ct * 16 + n];
}

__device__ __forceinline__ void store_cd_bf16(__bf16* Y, const v8f c[8]) {
    int lane = threadIdx.x & 31;
    int n = lane & 15, h = lane >> 4;
#pragma unroll
    for (int ct = 0; ct < 8; ++ct)
#pragma unroll
        for (int v = 0; v < 8; ++v)
            Y[(v + 8 * h) * DD + ct * 16 + n] = f2bf(c[ct][v]);
}

// GroupNorm over feature dim (128); rows distributed across 16-lane halves.
__device__ __forceinline__ void group_norm(v8f c[8], const float* gw, const float* gb) {
    int lane = threadIdx.x & 31;
    int n = lane & 15;
    float g[8], bb[8];
#pragma unroll
    for (int ct = 0; ct < 8; ++ct) { g[ct] = gw[ct * 16 + n]; bb[ct] = gb[ct * 16 + n]; }
#pragma unroll
    for (int v = 0; v < 8; ++v) {
        float s = 0.f, s2 = 0.f;
#pragma unroll
        for (int ct = 0; ct < 8; ++ct) { float x = c[ct][v]; s += x; s2 += x * x; }
#pragma unroll
        for (int off = 1; off < 16; off <<= 1) {
            s  += __shfl_xor(s,  off, 32);
            s2 += __shfl_xor(s2, off, 32);
        }
        float mu  = s  * (1.f / 128.f);
        float var = s2 * (1.f / 128.f) - mu * mu;
        float inv = rsqrtf(var + EPSV);
#pragma unroll
        for (int ct = 0; ct < 8; ++ct) c[ct][v] = (c[ct][v] - mu) * inv * g[ct] + bb[ct];
    }
}

__device__ __forceinline__ void relu8(v8f c[8]) {
#pragma unroll
    for (int ct = 0; ct < 8; ++ct)
#pragma unroll
        for (int v = 0; v < 8; ++v) c[ct][v] = fmaxf(c[ct][v], 0.f);
}

__device__ __forceinline__ void leaky8(v8f c[8]) {
#pragma unroll
    for (int ct = 0; ct < 8; ++ct)
#pragma unroll
        for (int v = 0; v < 8; ++v) {
            float x = c[ct][v];
            c[ct][v] = x > 0.f ? x : x * NEGS;
        }
}

// ---- Weight pre-pack: f32 (row-major, ld) -> bf16 B-operand tile layout ----
__global__ void __launch_bounds__(32)
pack_weight(const float* __restrict__ W, int ld, __bf16* __restrict__ P) {
    int lane = threadIdx.x & 31;
    int n = lane & 15, h = lane >> 4;
    int ct = blockIdx.x & 7, kc = blockIdx.x >> 3;   // grid = 32
    const float* src = W + (size_t)(ct * 16 + n) * ld + kc * 32 + h * 8;
    __bf16* dst = P + ((size_t)(kc * 8 + ct) * 32 + lane) * 16;
#pragma unroll
    for (int i = 0; i < 16; ++i) dst[i] = f2bf(src[i < 8 ? i : i + 8]);
}

// ---- Kernel 1: per-node precompute -----------------------------------------
__global__ void __launch_bounds__(32)
node_pre(const float* __restrict__ X,
         const __bf16* __restrict__ awp, const __bf16* __restrict__ qwp,
         const __bf16* __restrict__ cw1bp, const __bf16* __restrict__ cw1cp,
         const float* __restrict__ qgw, const float* __restrict__ qgb,
         float* __restrict__ ACC, float* __restrict__ QC, float* __restrict__ CC) {
    __shared__ __align__(16) __bf16 shA[16 * DD];
    __shared__ __align__(16) __bf16 shB[16 * DD];
    int r0 = blockIdx.x * 16;
    v8f acc[8];

    stage_rows(shA, X, DD, r0);
    __syncthreads();

    zero8(acc);
    gemm16p(shA, awp, acc);
    store_cd(ACC, DD, r0, acc);

    zero8(acc);
    gemm16p(shA, qwp, acc);
    group_norm(acc, qgw, qgb);
    relu8(acc);
    store_cd_bf16(shB, acc);
    __syncthreads();

    zero8(acc);
    gemm16p(shB, cw1bp, acc);
    store_cd(QC, DD, r0, acc);

    zero8(acc);
    gemm16p(shA, cw1cp, acc);
    store_cd(CC, DD, r0, acc);
}

// ---- Kernel 2: fused per-edge pipeline -------------------------------------
__global__ void __launch_bounds__(32)
edge_kernel(const float* __restrict__ ctrs,
            const int* __restrict__ hi, const int* __restrict__ wi, int E,
            const float* __restrict__ dw1, const float* __restrict__ db1,
            const __bf16* __restrict__ dw2p,
            const float* __restrict__ dgw, const float* __restrict__ dgb,
            const __bf16* __restrict__ cw1ap,
            const float* __restrict__ cgw, const float* __restrict__ cgb,
            const __bf16* __restrict__ cw2p,
            const float* __restrict__ QC, const float* __restrict__ CC,
            float* __restrict__ ACC) {
    __shared__ __align__(16) __bf16 sh[16 * DD];
    __shared__ int   shi[16], swi[16];
    __shared__ float sd0[16], sd1[16];

    int e0 = blockIdx.x * 16;
    int lane = threadIdx.x & 31;
    if (lane < 16) {
        int e = e0 + lane;
        int h = (e < E) ? hi[e] : 0;
        int w = (e < E) ? wi[e] : 0;
        shi[lane] = h; swi[lane] = w;
        sd0[lane] = ctrs[2 * h]     - ctrs[2 * w];
        sd1[lane] = ctrs[2 * h + 1] - ctrs[2 * w + 1];
    }
    __syncthreads();

    int n = lane & 15, hh = lane >> 4;

    // d2 = relu(d @ dw1^T + db1) built directly in C/D layout (K=2 only)
    v8f t[8];
#pragma unroll
    for (int ct = 0; ct < 8; ++ct) {
        int j = ct * 16 + n;
        float w0 = dw1[2 * j], w1 = dw1[2 * j + 1], b = db1[j];
#pragma unroll
        for (int v = 0; v < 8; ++v) {
            int m = v + 8 * hh;
            t[ct][v] = fmaxf(sd0[m] * w0 + sd1[m] * w1 + b, 0.f);
        }
    }
    store_cd_bf16(sh, t);
    __syncthreads();

    // d3 = relu(gn(d2 @ dw2^T))
    v8f acc[8];
    zero8(acc);
    gemm16p(sh, dw2p, acc);
    group_norm(acc, dgw, dgb);
    relu8(acc);
    __syncthreads();
    store_cd_bf16(sh, acc);
    __syncthreads();

    // e = d3 @ cw1a^T + QC[hi] + CC[wi]
#pragma unroll
    for (int ct = 0; ct < 8; ++ct)
#pragma unroll
        for (int v = 0; v < 8; ++v) {
            int m = v + 8 * hh;
            acc[ct][v] = QC[(size_t)shi[m] * DD + ct * 16 + n]
                       + CC[(size_t)swi[m] * DD + ct * 16 + n];
        }
    gemm16p(sh, cw1ap, acc);
    group_norm(acc, cgw, cgb);
    relu8(acc);
    __syncthreads();
    store_cd_bf16(sh, acc);
    __syncthreads();

    // o = e @ cw2^T ; scatter-add into ACC[hi]
    zero8(acc);
    gemm16p(sh, cw2p, acc);
#pragma unroll
    for (int ct = 0; ct < 8; ++ct)
#pragma unroll
        for (int v = 0; v < 8; ++v) {
            int m = v + 8 * hh;
            if (e0 + m < E)
                atomicAdd(&ACC[(size_t)shi[m] * DD + ct * 16 + n], acc[ct][v]);
        }
}

// ---- Kernel 3: per-node epilogue -------------------------------------------
__global__ void __launch_bounds__(32)
node_post(const float* __restrict__ X, const float* __restrict__ ACC,
          const float* __restrict__ nw, const float* __restrict__ nb,
          const __bf16* __restrict__ lwp,
          const float* __restrict__ lgw, const float* __restrict__ lgb,
          float* __restrict__ OUT) {
    __shared__ __align__(16) __bf16 sh[16 * DD];
    int r0 = blockIdx.x * 16;

    v8f a[8];
    load_cd(ACC, DD, r0, a);
    group_norm(a, nw, nb);
    leaky8(a);
    store_cd_bf16(sh, a);
    __syncthreads();

    v8f t[8];
    zero8(t);
    gemm16p(sh, lwp, t);
    group_norm(t, lgw, lgb);

    v8f r[8];
    load_cd(X, DD, r0, r);
#pragma unroll
    for (int ct = 0; ct < 8; ++ct)
#pragma unroll
        for (int v = 0; v < 8; ++v) {
            float x = t[ct][v] + r[ct][v];
            t[ct][v] = x > 0.f ? x : x * NEGS;
        }
    store_cd(OUT, DD, r0, t);
}

extern "C" void kernel_launch(void* const* d_in, const int* in_sizes, int n_in,
                              void* d_out, int out_size, void* d_ws, size_t ws_size,
                              hipStream_t stream) {
    const float* actors   = (const float*)d_in[0];
    const float* ctrs     = (const float*)d_in[1];
    const int*   hi       = (const int*)d_in[2];
    const int*   wi       = (const int*)d_in[3];
    const float* dist_w1  = (const float*)d_in[4];
    const float* dist_b1  = (const float*)d_in[5];
    const float* dist_w2  = (const float*)d_in[6];
    const float* dist_gw  = (const float*)d_in[7];
    const float* dist_gb  = (const float*)d_in[8];
    const float* query_w  = (const float*)d_in[9];
    const float* query_gw = (const float*)d_in[10];
    const float* query_gb = (const float*)d_in[11];
    const float* ctx_w1   = (const float*)d_in[12];
    const float* ctx_gw   = (const float*)d_in[13];
    const float* ctx_gb   = (const float*)d_in[14];
    const float* ctx_w2   = (const float*)d_in[15];
    const float* agt_w    = (const float*)d_in[16];
    const float* norm_w   = (const float*)d_in[17];
    const float* norm_b   = (const float*)d_in[18];
    const float* lin_w    = (const float*)d_in[19];
    const float* lin_gw   = (const float*)d_in[20];
    const float* lin_gb   = (const float*)d_in[21];

    const int E = in_sizes[2];
    const int N = in_sizes[0] / DD;

    float* ACC = (float*)d_ws;
    float* QC  = ACC + (size_t)N * DD;
    float* CC  = QC  + (size_t)N * DD;
    float* MID = CC  + (size_t)N * DD;
    __bf16* WP = (__bf16*)(MID + (size_t)N * DD);   // 16 packed 128x128 bf16 matrices

    const size_t MSZ = (size_t)DD * DD;
    // matrix slots per layer: 0=aw 1=qw 2=cw1a 3=cw1b 4=cw1c 5=dw2 6=cw2 7=lw
    for (int i = 0; i < 2; ++i) {
        size_t oDD = (size_t)i * DD * DD;
        size_t oD3 = (size_t)i * DD * 3 * DD;
        __bf16* base = WP + (size_t)i * 8 * MSZ;
        hipLaunchKernelGGL(pack_weight, dim3(32), dim3(32), 0, stream, agt_w  + oDD, DD,     base + 0 * MSZ);
        hipLaunchKernelGGL(pack_weight, dim3(32), dim3(32), 0, stream, query_w+ oDD, DD,     base + 1 * MSZ);
        hipLaunchKernelGGL(pack_weight, dim3(32), dim3(32), 0, stream, ctx_w1 + oD3,          3 * DD, base + 2 * MSZ);
        hipLaunchKernelGGL(pack_weight, dim3(32), dim3(32), 0, stream, ctx_w1 + oD3 + DD,     3 * DD, base + 3 * MSZ);
        hipLaunchKernelGGL(pack_weight, dim3(32), dim3(32), 0, stream, ctx_w1 + oD3 + 2 * DD, 3 * DD, base + 4 * MSZ);
        hipLaunchKernelGGL(pack_weight, dim3(32), dim3(32), 0, stream, dist_w2+ oDD, DD,     base + 5 * MSZ);
        hipLaunchKernelGGL(pack_weight, dim3(32), dim3(32), 0, stream, ctx_w2 + oDD, DD,     base + 6 * MSZ);
        hipLaunchKernelGGL(pack_weight, dim3(32), dim3(32), 0, stream, lin_w  + oDD, DD,     base + 7 * MSZ);
    }

    const float* act = actors;
    const int ntiles = N / 16;
    const int etiles = (E + 15) / 16;

    for (int i = 0; i < 2; ++i) {
        size_t oD  = (size_t)i * DD;
        size_t oD2 = (size_t)i * DD * 2;
        __bf16* base = WP + (size_t)i * 8 * MSZ;
        float* out_act = (i == 0) ? MID : (float*)d_out;

        hipLaunchKernelGGL(node_pre, dim3(ntiles), dim3(32), 0, stream,
                           act, base + 0 * MSZ, base + 1 * MSZ,
                           base + 3 * MSZ, base + 4 * MSZ,
                           query_gw + oD, query_gb + oD,
                           ACC, QC, CC);

        hipLaunchKernelGGL(edge_kernel, dim3(etiles), dim3(32), 0, stream,
                           ctrs, hi, wi, E,
                           dist_w1 + oD2, dist_b1 + oD, base + 5 * MSZ,
                           dist_gw + oD, dist_gb + oD,
                           base + 2 * MSZ, ctx_gw + oD, ctx_gb + oD,
                           base + 6 * MSZ, QC, CC, ACC);

        hipLaunchKernelGGL(node_post, dim3(ntiles), dim3(32), 0, stream,
                           act, ACC, norm_w + oD, norm_b + oD,
                           base + 7 * MSZ, lin_gw + oD, lin_gb + oD, out_act);

        act = out_act;
    }
}
